// TitansMemory_76845554860056
// MI455X (gfx1250) — matrix operand order repaired and checked by reading, not compiled, not run on hardware
//
#include <hip/hip_runtime.h>
#include <math.h>

// ---------------- types ----------------
typedef __attribute__((ext_vector_type(16))) __bf16 v16bf;
typedef __attribute__((ext_vector_type(8)))  float  v8f;
typedef __attribute__((ext_vector_type(4)))  unsigned int u32x4;

union FragBF { u32x4 h[2]; v16bf v; };

__device__ __forceinline__ unsigned short f2bf(float f) {
    unsigned int u = __float_as_uint(f);
    u += 0x7fffu + ((u >> 16) & 1u);   // RNE to bf16
    return (unsigned short)(u >> 16);
}

#define BM 128
#define BN 128
#define BK 32
#define BKP 40   // padded LDS row (ushorts): 80B stride, keeps 16B alignment

// ---------------- WMMA GEMM: bf16 operands, fp32 accumulate ----------------
// A: [Mr][Kd] bf16 row-major, lda in elements.  Bw: [Nc][Kd] bf16 row-major.
// C[m,n] = act( alpha*sA * sum_k A[m,k]*Bw[n,k] + bias[n] ) + cA*addA + cB*addB
__global__ __launch_bounds__(256)
void titans_wmma_gemm_bf16(const unsigned short* __restrict__ A, int lda,
                           const unsigned short* __restrict__ Bw, int ldb,
                           float* __restrict__ C, unsigned short* __restrict__ Cb, int ldc,
                           int Mr, int Nc, int Kd,
                           float alpha, const float* sAlpha,
                           const float* bias, int act,
                           const float* addA, float cA, const float* sCA,
                           const float* addB, float cB, const float* sCB)
{
    __shared__ __align__(16) unsigned short At[2][BM][BKP];
    __shared__ __align__(16) unsigned short Bt[2][BN][BKP];

    const int tid  = threadIdx.x;
    const int wave = tid >> 5;
    const int lane = tid & 31;
    const int waveM = wave >> 2;         // 0..1 : 64 rows
    const int waveN = wave & 3;          // 0..3 : 32 cols
    const int m0 = blockIdx.y * BM;
    const int n0 = blockIdx.x * BN;

    // stage one BMxBK bf16 tile into LDS.
    // fast path: async global->LDS, 16B per lane, 2 instructions per wave.
    auto stage = [&](const unsigned short* __restrict__ G, int ld, int r0, int rows,
                     int k0, unsigned short (*T)[BKP], bool full) {
        if (full) {
            #pragma unroll
            for (int j = 0; j < 2; j++) {
                int row = (wave * 2 + j) * 8 + (lane >> 2);
                int seg = (lane & 3) * 8;
                unsigned ldsa = (unsigned)(size_t)&T[row][seg];
                unsigned long long ga =
                    (unsigned long long)(size_t)(G + (size_t)(r0 + row) * ld + k0 + seg);
                asm volatile("global_load_async_to_lds_b128 %0, %1, off"
                             :: "v"(ldsa), "v"(ga) : "memory");
            }
        } else {
            int srow = tid >> 1;
            int sks  = (tid & 1) * 16;
            int g = r0 + srow;
            #pragma unroll
            for (int i = 0; i < 16; i++) {
                int gk = k0 + sks + i;
                T[srow][sks + i] = (g < rows && gk < Kd)
                                 ? G[(size_t)g * ld + gk] : (unsigned short)0;
            }
        }
    };

    v8f acc[4][2];
    #pragma unroll
    for (int i = 0; i < 4; i++)
        #pragma unroll
        for (int j = 0; j < 2; j++)
            #pragma unroll
            for (int e = 0; e < 8; e++) acc[i][j][e] = 0.0f;

    const int nt = (Kd + BK - 1) / BK;
    const bool fullM = (m0 + BM) <= Mr;
    const bool fullN = (n0 + BN) <= Nc;

    stage(A,  lda, m0, Mr, 0, At[0], fullM && BK <= Kd);
    stage(Bw, ldb, n0, Nc, 0, Bt[0], fullN && BK <= Kd);
    asm volatile("s_wait_asynccnt 0x0" ::: "memory");
    __syncthreads();

    int cur = 0;
    for (int i = 0; i < nt; i++) {
        const bool hasNext = (i + 1) < nt;
        if (hasNext) {  // stage next tile into back buffer while we compute
            int k0n = (i + 1) * BK;
            bool fk = (k0n + BK) <= Kd;
            stage(A,  lda, m0, Mr, k0n, At[1 - cur], fullM && fk);
            stage(Bw, ldb, n0, Nc, k0n, Bt[1 - cur], fullN && fk);
        }
        // ---- compute on buffer `cur` ----
        FragBF af[4];
        #pragma unroll
        for (int sm = 0; sm < 4; sm++) {
            int r  = waveM * 64 + sm * 16 + (lane & 15);
            int kh = (lane < 16) ? 0 : 8;     // K halves {0..7,16..23} / {8..15,24..31}
            af[sm].h[0] = *(const u32x4*)&At[cur][r][kh];
            af[sm].h[1] = *(const u32x4*)&At[cur][r][kh + 16];
        }
        FragBF bfr[2];
        #pragma unroll
        for (int sn = 0; sn < 2; sn++) {
            int n  = waveN * 32 + sn * 16 + (lane & 15);
            int kh = (lane < 16) ? 0 : 8;     // K halves {0..7,16..23} / {8..15,24..31}
            bfr[sn].h[0] = *(const u32x4*)&Bt[cur][n][kh];
            bfr[sn].h[1] = *(const u32x4*)&Bt[cur][n][kh + 16];
        }
        #pragma unroll
        for (int sm = 0; sm < 4; sm++)
            #pragma unroll
            for (int sn = 0; sn < 2; sn++)
                acc[sm][sn] = __builtin_amdgcn_wmma_f32_16x16x32_bf16(
                    false, af[sm].v, false, bfr[sn].v,
                    (short)0, acc[sm][sn], false, false);

        if (hasNext) {
            asm volatile("s_wait_asynccnt 0x0" ::: "memory"); // own async writes done
            __syncthreads();                                  // publish back buffer
            cur ^= 1;
        }
    }

    // ---- epilogue ----
    float sA  = alpha * (sAlpha ? *sAlpha : 1.0f);
    float scA = cA    * (sCA    ? *sCA    : 1.0f);
    float scB = cB    * (sCB    ? *sCB    : 1.0f);

    #pragma unroll
    for (int sm = 0; sm < 4; sm++) {
        #pragma unroll
        for (int sn = 0; sn < 2; sn++) {
            int gn = n0 + waveN * 32 + sn * 16 + (lane & 15);
            if (gn >= Nc) continue;
            #pragma unroll
            for (int r = 0; r < 8; r++) {
                int gm = m0 + waveM * 64 + sm * 16 + r + ((lane < 16) ? 0 : 8);
                if (gm >= Mr) continue;
                float vv = sA * acc[sm][sn][r];
                if (bias) vv += bias[gn];
                if (act == 1)      vv = vv / (1.0f + __expf(-vv));   // silu
                else if (act == 2) vv = 1.0f / (1.0f + __expf(-vv)); // sigmoid
                size_t idx = (size_t)gm * ldc + gn;
                if (addA) vv += scA * addA[idx];
                if (addB) vv += scB * addB[idx];
                if (C)  C[idx]  = vv;
                if (Cb) Cb[idx] = f2bf(vv);
            }
        }
    }
}

// ---------------- fp32 -> bf16 flat cast ----------------
__global__ void titans_cast(unsigned short* __restrict__ dst,
                            const float* __restrict__ src, int n)
{ int i = blockIdx.x * blockDim.x + threadIdx.x; if (i < n) dst[i] = f2bf(src[i]); }

// ---------------- LDS-tiled transpose casts: dst[C][R] <- src[R][C] ----------------
#define TT 32
__global__ void titans_castT(unsigned short* __restrict__ dst,
                             const float* __restrict__ src, int R, int Ccols)
{
    __shared__ unsigned short t[TT][TT + 1];
    int c0 = blockIdx.x * TT, r0 = blockIdx.y * TT;
    for (int j = threadIdx.y; j < TT; j += blockDim.y) {
        int r = r0 + j, c = c0 + threadIdx.x;
        t[j][threadIdx.x] = (r < R && c < Ccols)
                          ? f2bf(src[(size_t)r * Ccols + c]) : (unsigned short)0;
    }
    __syncthreads();
    for (int j = threadIdx.y; j < TT; j += blockDim.y) {
        int c = c0 + j, r = r0 + threadIdx.x;
        if (c < Ccols && r < R) dst[(size_t)c * R + r] = t[threadIdx.x][j];
    }
}
__global__ void titans_castT_b(unsigned short* __restrict__ dst,
                               const unsigned short* __restrict__ src, int R, int Ccols)
{
    __shared__ unsigned short t[TT][TT + 1];
    int c0 = blockIdx.x * TT, r0 = blockIdx.y * TT;
    for (int j = threadIdx.y; j < TT; j += blockDim.y) {
        int r = r0 + j, c = c0 + threadIdx.x;
        t[j][threadIdx.x] = (r < R && c < Ccols)
                          ? src[(size_t)r * Ccols + c] : (unsigned short)0;
    }
    __syncthreads();
    for (int j = threadIdx.y; j < TT; j += blockDim.y) {
        int c = c0 + j, r = r0 + threadIdx.x;
        if (c < Ccols && r < R) dst[(size_t)c * R + r] = t[threadIdx.x][j];
    }
}

// ---------------- depthwise causal conv (bf16 out) ----------------
__global__ void titans_dwconv(const float* __restrict__ in, const float* __restrict__ w,
                              unsigned short* __restrict__ out, int total, int L, int Mch, int K)
{
    int i = blockIdx.x * blockDim.x + threadIdx.x;
    if (i >= total) return;
    int c = i % Mch;
    int t = i / Mch;
    int l = t % L;
    float s = 0.0f;
    for (int j = 0; j < K; j++) {
        int ls = l - (K - 1) + j;
        if (ls >= 0) s += w[c * K + j] * in[(size_t)(t - (K - 1) + j) * Mch + c];
    }
    out[i] = f2bf(s);
}

// ---------------- row L2 normalize (bf16 out) ----------------
__global__ void titans_l2norm_b(unsigned short* __restrict__ dst,
                                const float* __restrict__ src, int cols)
{
    int row = blockIdx.x;
    const float* r = src + (size_t)row * cols;
    unsigned short* o = dst + (size_t)row * cols;
    __shared__ float red[256];
    float s = 0.0f;
    for (int c = threadIdx.x; c < cols; c += blockDim.x) { float v = r[c]; s += v * v; }
    red[threadIdx.x] = s; __syncthreads();
    for (int st = 128; st > 0; st >>= 1) {
        if (threadIdx.x < st) red[threadIdx.x] += red[threadIdx.x + st];
        __syncthreads();
    }
    float inv = 1.0f / (sqrtf(red[0]) + 1e-6f);
    for (int c = threadIdx.x; c < cols; c += blockDim.x) o[c] = f2bf(r[c] * inv);
}

// ---------------- causal softmax -> bf16 probabilities ----------------
__global__ void titans_softmax(float* __restrict__ sc, unsigned short* __restrict__ outb,
                               int cols, int P, int L, float scale)
{
    int row = blockIdx.x;
    float* r = sc + (size_t)row * cols;
    unsigned short* o = outb + (size_t)row * cols;
    int limit = P + (row % L);           // inclusive max valid column
    __shared__ float red[256];
    float mx = -3.4e38f;
    for (int c = threadIdx.x; c <= limit; c += blockDim.x) mx = fmaxf(mx, r[c] * scale);
    red[threadIdx.x] = mx; __syncthreads();
    for (int st = 128; st > 0; st >>= 1) {
        if (threadIdx.x < st) red[threadIdx.x] = fmaxf(red[threadIdx.x], red[threadIdx.x + st]);
        __syncthreads();
    }
    mx = red[0]; __syncthreads();
    float sum = 0.0f;
    for (int c = threadIdx.x; c <= limit; c += blockDim.x) {
        float e = __expf(r[c] * scale - mx);
        r[c] = e; sum += e;
    }
    red[threadIdx.x] = sum; __syncthreads();
    for (int st = 128; st > 0; st >>= 1) {
        if (threadIdx.x < st) red[threadIdx.x] += red[threadIdx.x + st];
        __syncthreads();
    }
    float inv = 1.0f / red[0];
    for (int c = threadIdx.x; c <= limit; c += blockDim.x) o[c] = f2bf(r[c] * inv);
    for (int c = limit + 1 + threadIdx.x; c < cols; c += blockDim.x) o[c] = 0;
}

// ---------------- elementwise ----------------
__global__ void titans_ew_add(float* c, const float* a, const float* b, int n)
{ int i = blockIdx.x * blockDim.x + threadIdx.x; if (i < n) c[i] = a[i] + b[i]; }
__global__ void titans_ew_mul(float* c, const float* a, const float* b, int n)
{ int i = blockIdx.x * blockDim.x + threadIdx.x; if (i < n) c[i] = a[i] * b[i]; }

// ---------------- pooled scalar gates eta/theta/alpha ----------------
__global__ void titans_pool_dot(const float* __restrict__ x,
                                const float* __restrict__ ew,
                                const float* __restrict__ tw,
                                const float* __restrict__ aw,
                                float* __restrict__ acc, int total, int LD, int D)
{
    float la[6] = {0, 0, 0, 0, 0, 0};
    for (int i = blockIdx.x * blockDim.x + threadIdx.x; i < total;
         i += gridDim.x * blockDim.x) {
        int d = i % D;
        int b = i / LD;
        float v = x[i];
        la[b * 3 + 0] += v * ew[d];
        la[b * 3 + 1] += v * tw[d];
        la[b * 3 + 2] += v * aw[d];
    }
    for (int j = 0; j < 6; j++) atomicAdd(&acc[j], la[j]);
}

__global__ void titans_pool_fin(const float* acc, const float* eb, const float* tb,
                                const float* ab, float* s, int B, int L)
{
    if (blockIdx.x == 0 && threadIdx.x == 0) {
        float eta = 0.0f, th = 0.0f, al = 0.0f;
        for (int b = 0; b < B; b++) {
            eta += 1.0f / (1.0f + __expf(-(acc[b * 3 + 0] / (float)L + eb[0])));
            th  += 1.0f / (1.0f + __expf(-(acc[b * 3 + 1] / (float)L + tb[0])));
            al  += 1.0f / (1.0f + __expf(-(acc[b * 3 + 2] / (float)L + ab[0])));
        }
        s[0] = eta / (float)B;
        s[1] = th  / (float)B;
        s[2] = 1.0f - al / (float)B;
    }
}

// ---------------- RMSNorm ----------------
__global__ void titans_rmsnorm(float* __restrict__ out, const float* __restrict__ in,
                               const float* __restrict__ w, int cols)
{
    int row = blockIdx.x;
    const float* r = in + (size_t)row * cols;
    float* o = out + (size_t)row * cols;
    __shared__ float red[256];
    float s = 0.0f;
    for (int c = threadIdx.x; c < cols; c += blockDim.x) { float v = r[c]; s += v * v; }
    red[threadIdx.x] = s; __syncthreads();
    for (int st = 128; st > 0; st >>= 1) {
        if (threadIdx.x < st) red[threadIdx.x] += red[threadIdx.x + st];
        __syncthreads();
    }
    float inv = 1.0f / sqrtf(red[0] / (float)cols + 1e-6f);
    for (int c = threadIdx.x; c < cols; c += blockDim.x) o[c] = r[c] * inv * w[c];
}

// ---------------- host helpers ----------------
static void g_gemm(hipStream_t s,
                   const unsigned short* A, int lda,
                   const unsigned short* Bw, int ldb,
                   float* C, unsigned short* Cb, int ldc,
                   int Mr, int Nc, int Kd,
                   float alpha = 1.0f, const float* sAlpha = nullptr,
                   const float* bias = nullptr, int act = 0,
                   const float* addA = nullptr, float cA = 0.0f, const float* sCA = nullptr,
                   const float* addB = nullptr, float cB = 0.0f, const float* sCB = nullptr)
{
    dim3 grid((Nc + BN - 1) / BN, (Mr + BM - 1) / BM);
    titans_wmma_gemm_bf16<<<grid, dim3(256), 0, s>>>(A, lda, Bw, ldb, C, Cb, ldc,
        Mr, Nc, Kd, alpha, sAlpha, bias, act, addA, cA, sCA, addB, cB, sCB);
}
static void g_cast(hipStream_t s, unsigned short* d, const float* f, int n)
{ titans_cast<<<(n + 255) / 256, 256, 0, s>>>(d, f, n); }
static void g_castT(hipStream_t s, unsigned short* d, const float* f, int R, int Ccols)
{ titans_castT<<<dim3((Ccols + TT - 1) / TT, (R + TT - 1) / TT), dim3(TT, 8), 0, s>>>(d, f, R, Ccols); }
static void g_castT_b(hipStream_t s, unsigned short* d, const unsigned short* f, int R, int Ccols)
{ titans_castT_b<<<dim3((Ccols + TT - 1) / TT, (R + TT - 1) / TT), dim3(TT, 8), 0, s>>>(d, f, R, Ccols); }

extern "C" void kernel_launch(void* const* d_in, const int* in_sizes, int n_in,
                              void* d_out, int out_size, void* d_ws, size_t ws_size,
                              hipStream_t stream)
{
    const int B = 2, L = 1024, D = 2048, M = 2048, P = 8, KC = 4;
    const int BL = B * L;                    // 2048
    const int BLM = BL * M;                  // 4,194,304
    const int COLS = P + L;                  // 1032

    const float* x      = (const float*)d_in[0];
    const float* Wq     = (const float*)d_in[1];
    const float* Wk     = (const float*)d_in[2];
    const float* Wv     = (const float*)d_in[3];
    const float* q_dw   = (const float*)d_in[4];
    const float* k_dw   = (const float*)d_in[5];
    const float* v_dw   = (const float*)d_in[6];
    const float* q_pw   = (const float*)d_in[7];
    const float* k_pw   = (const float*)d_in[8];
    const float* v_pw   = (const float*)d_in[9];
    const float* pk     = (const float*)d_in[10];
    const float* pv     = (const float*)d_in[11];
    const float* mlp_w1 = (const float*)d_in[12];
    const float* mlp_w2 = (const float*)d_in[13];
    const float* eta_w  = (const float*)d_in[14];
    const float* eta_b  = (const float*)d_in[15];
    const float* theta_w= (const float*)d_in[16];
    const float* theta_b= (const float*)d_in[17];
    const float* alpha_w= (const float*)d_in[18];
    const float* alpha_b= (const float*)d_in[19];
    const float* gate_w = (const float*)d_in[20];
    const float* gate_b = (const float*)d_in[21];
    const float* out_w  = (const float*)d_in[22];
    const float* ln_w   = (const float*)d_in[23];
    const float* mem    = (const float*)d_in[24];
    const float* sur    = (const float*)d_in[25];

    // ---- fp32 workspace ----
    float* wsf  = (float*)d_ws;
    float* t1   = wsf;  wsf += BLM;
    float* t2   = wsf;  wsf += BLM;
    float* vF   = wsf;  wsf += BLM;
    float* snap = wsf;  wsf += (size_t)M * M;
    float* sc   = wsf;  wsf += (size_t)BL * COLS;
    float* acc  = wsf;  wsf += 8;
    float* svec = wsf;  wsf += 8;   // [eta, theta, 1-alpha]
    // ---- bf16 workspace ----
    unsigned short* wsb = (unsigned short*)wsf;
    unsigned short* xb   = wsb;  wsb += (size_t)BL * D;
    unsigned short* wb   = wsb;  wsb += (size_t)M * D;   // rotating weight slot
    unsigned short* adb  = wsb;  wsb += BLM;             // rotating activation slot
    unsigned short* qb   = wsb;  wsb += BLM;
    unsigned short* kbB  = wsb;  wsb += BLM;
    unsigned short* tb   = wsb;  wsb += BLM;             // kT then vT
    unsigned short* mtb  = wsb;  wsb += (size_t)M * M;   // memT -> errT -> snapT
    unsigned short* scb  = wsb;  wsb += (size_t)BL * COLS;
    unsigned short* pknb = wsb;  wsb += (size_t)P * M;
    unsigned short* pvTb = wsb;  wsb += (size_t)M * P;

    const int EB = 256;
    const int BLK_BLM = (BLM + EB - 1) / EB;

    // data-dependent scalars
    hipMemsetAsync(acc, 0, 6 * sizeof(float), stream);
    titans_pool_dot<<<256, EB, 0, stream>>>(x, eta_w, theta_w, alpha_w, acc,
                                            B * L * D, L * D, D);
    titans_pool_fin<<<1, 32, 0, stream>>>(acc, eta_b, theta_b, alpha_b, svec, B, L);

    // one-time conversions
    g_cast(stream, xb, x, BL * D);
    titans_l2norm_b<<<P, EB, 0, stream>>>(pknb, pk, M);          // pk_n (bf16)
    g_castT(stream, pvTb, pv, P, M);                             // pv^T [M][P]

    // q = l2n(silu(dwconv(x @ Wq.T) @ q_pw.T))  -> qb (bf16)
    g_cast(stream, wb, Wq, M * D);
    g_gemm(stream, xb, D, wb, D, t1, nullptr, M, BL, M, D);
    titans_dwconv<<<BLK_BLM, EB, 0, stream>>>(t1, q_dw, adb, BLM, L, M, KC);
    g_cast(stream, wb, q_pw, M * M);
    g_gemm(stream, adb, M, wb, M, t1, nullptr, M, BL, M, M, 1.0f, nullptr, nullptr, 1);
    titans_l2norm_b<<<BL, EB, 0, stream>>>(qb, t1, M);
    // k -> kbB (bf16), k^T -> tb
    g_cast(stream, wb, Wk, M * D);
    g_gemm(stream, xb, D, wb, D, t1, nullptr, M, BL, M, D);
    titans_dwconv<<<BLK_BLM, EB, 0, stream>>>(t1, k_dw, adb, BLM, L, M, KC);
    g_cast(stream, wb, k_pw, M * M);
    g_gemm(stream, adb, M, wb, M, t1, nullptr, M, BL, M, M, 1.0f, nullptr, nullptr, 1);
    titans_l2norm_b<<<BL, EB, 0, stream>>>(kbB, t1, M);
    g_castT_b(stream, tb, kbB, BL, M);                           // kT [M][BL]
    // v -> vF (fp32, needed as addend)
    g_cast(stream, wb, Wv, M * D);
    g_gemm(stream, xb, D, wb, D, t1, nullptr, M, BL, M, D);
    titans_dwconv<<<BLK_BLM, EB, 0, stream>>>(t1, v_dw, adb, BLM, L, M, KC);
    g_cast(stream, wb, v_pw, M * M);
    g_gemm(stream, adb, M, wb, M, vF, nullptr, M, BL, M, M, 1.0f, nullptr, nullptr, 1);

    // err = v - k @ mem  -> t1 (fp32)
    g_castT(stream, mtb, mem, M, M);                             // mem^T
    g_gemm(stream, kbB, M, mtb, M, t1, nullptr, M, BL, M, M,
           -1.0f, nullptr, nullptr, 0, vF, 1.0f, nullptr);
    // snapshot = (1-alpha)*mem + eta*sur + theta*(k^T err)/(B*L)
    g_castT(stream, mtb, t1, BL, M);                             // err^T [M][BL]
    g_gemm(stream, tb, BL, mtb, BL, snap, nullptr, M, M, M, BL,
           1.0f / (float)BL, svec + 1, nullptr, 0,
           sur, 1.0f, svec + 0, mem, 1.0f, svec + 2);

    // linear retrieval: t1 = q @ snap
    g_castT(stream, mtb, snap, M, M);                            // snap^T
    g_gemm(stream, qb, M, mtb, M, t1, nullptr, M, BL, M, M);
    // mlp: adb = silu(q @ w1.T) (bf16 direct);  t1 += adb @ w2.T
    g_cast(stream, wb, mlp_w1, M * M);
    g_gemm(stream, qb, M, wb, M, nullptr, adb, M, BL, M, M, 1.0f, nullptr, nullptr, 1);
    g_cast(stream, wb, mlp_w2, M * M);
    g_gemm(stream, adb, M, wb, M, t1, nullptr, M, BL, M, M,
           1.0f, nullptr, nullptr, 0, t1, 1.0f, nullptr);

    // attention scores
    g_gemm(stream, qb, M, pknb, M, sc, nullptr, COLS, BL, P, M);
    for (int b = 0; b < B; b++)
        g_gemm(stream, qb + (size_t)b * L * M, M, kbB + (size_t)b * L * M, M,
               sc + (size_t)b * L * COLS + P, nullptr, COLS, L, L, M);
    titans_softmax<<<BL, EB, 0, stream>>>(sc, scb, COLS, P, L, 1.0f / sqrtf((float)M));

    // context: t2 = attn[:, :P] @ pv ; then += attn_b[:, P:] @ v_b
    g_castT(stream, tb, vF, BL, M);                              // v^T [M][BL]
    g_gemm(stream, scb, COLS, pvTb, P, t2, nullptr, M, BL, M, P);
    for (int b = 0; b < B; b++)
        g_gemm(stream, scb + (size_t)b * L * COLS + P, COLS,
               tb + (size_t)b * L, BL,
               t2 + (size_t)b * L * M, nullptr, M, L, M, L,
               1.0f, nullptr, nullptr, 0,
               t2 + (size_t)b * L * M, 1.0f, nullptr);

    // retrieved = t1 + t2 ; bf16 copy for GEMM A
    titans_ew_add<<<BLK_BLM, EB, 0, stream>>>(t1, t1, t2, BLM);
    g_cast(stream, adb, t1, BLM);
    // gate = sigmoid(retrieved @ gate_w.T + gate_b) -> t2
    g_cast(stream, wb, gate_w, M * M);
    g_gemm(stream, adb, M, wb, M, t2, nullptr, M, BL, M, M,
           1.0f, nullptr, gate_b, 2);
    // t2 = gate * retrieved ; bf16 copy
    titans_ew_mul<<<BLK_BLM, EB, 0, stream>>>(t2, t2, t1, BLM);
    g_cast(stream, adb, t2, BLM);
    // out_pre = x + (gate*retrieved) @ out_w.T -> t1 (BL x D)
    g_cast(stream, wb, out_w, D * M);
    g_gemm(stream, adb, M, wb, M, t1, nullptr, D, BL, D, M,
           1.0f, nullptr, nullptr, 0, x, 1.0f, nullptr);
    // RMSNorm -> d_out
    titans_rmsnorm<<<BL, EB, 0, stream>>>((float*)d_out, t1, ln_w, D);
}
